// InvertibleResBlock_55997783605825
// MI455X (gfx1250) — matrix operand order repaired and checked by reading, not compiled, not run on hardware
//
#include <hip/hip_runtime.h>
#include <hip/hip_bf16.h>
#include <hip/hip_fp16.h>
#include <math.h>

typedef __attribute__((ext_vector_type(16))) _Float16 v16h;
typedef __attribute__((ext_vector_type(8)))  _Float16 v8h;
typedef __attribute__((ext_vector_type(8)))  float    v8f;

#define BATCH   262144
#define DIN     64
#define HID     128
#define NTERMS  8
#define NPOW    5

// Packed-weight workspace layout (in _Float16 units). Weights are stored as
// WMMA *A-operand* fragments of W itself (rows = output features):
//   blob(mt,kc), idx = mt*kcCount + kc, 512 halves each, lane l holds 16
//   contiguous halves: m = mt*16 + (l&15), k = kc*32 + (l>>4)*16 + {0..15}
//   W0 (K=64 ,M=128): 16 blobs at off 0
//   W1 (K=128,M=128): 32 blobs at off 8192
//   W2 (K=128,M=64 ): 16 blobs at off 24576
#define W0_OFF 0
#define W1_OFF 8192
#define W2_OFF 24576
#define WPACK_HALFS 32768

// ---------------------------------------------------------------------------
// Kernel 1: spectral normalization (power iteration) + pack to A-frag layout
// ---------------------------------------------------------------------------
__device__ inline float block_reduce_sum_128(float x, float* red) {
  int t = threadIdx.x;
  red[t] = x;
  __syncthreads();
  #pragma unroll
  for (int s = 64; s > 0; s >>= 1) {
    if (t < s) red[t] += red[t + s];
    __syncthreads();
  }
  float r = red[0];
  __syncthreads();
  return r;
}

__device__ float spectral_sigma(const float* W, const float* uin,
                                int rows, int cols,
                                float* u, float* v, float* red) {
  int t = threadIdx.x;
  if (t < rows) u[t] = uin[t];
  __syncthreads();
  for (int it = 0; it < NPOW; ++it) {
    float s = 0.0f;
    if (t < cols) {
      for (int i = 0; i < rows; ++i) s += W[i * cols + t] * u[i];
    }
    float n2 = block_reduce_sum_128((t < cols) ? s * s : 0.0f, red);
    float inv = 1.0f / (sqrtf(n2) + 1e-12f);
    if (t < cols) v[t] = s * inv;
    __syncthreads();
    float s2 = 0.0f;
    if (t < rows) {
      for (int j = 0; j < cols; ++j) s2 += W[t * cols + j] * v[j];
    }
    float nu2 = block_reduce_sum_128((t < rows) ? s2 * s2 : 0.0f, red);
    float invu = 1.0f / (sqrtf(nu2) + 1e-12f);
    if (t < rows) u[t] = s2 * invu;
    __syncthreads();
  }
  float s3 = 0.0f;
  if (t < rows) {
    for (int j = 0; j < cols; ++j) s3 += W[t * cols + j] * v[j];
    s3 *= u[t];
  }
  return block_reduce_sum_128((t < rows) ? s3 : 0.0f, red);
}

__device__ void pack_afrags(const float* W, float inv_sigma,
                            int kdim, int mdim, _Float16* dst) {
  int kc_count = kdim >> 5;
  int total = (mdim >> 4) * kc_count * 512;
  for (int idx = threadIdx.x; idx < total; idx += blockDim.x) {
    int blob   = idx >> 9;
    int within = idx & 511;
    int l  = within >> 4;
    int e  = within & 15;
    int mt = blob / kc_count;
    int kc = blob - mt * kc_count;
    int k  = kc * 32 + ((l >> 4) << 4) + e;
    int m  = mt * 16 + (l & 15);
    dst[idx] = (_Float16)(W[m * kdim + k] * inv_sigma);
  }
}

__global__ __launch_bounds__(128) void sn_pack_kernel(
    const float* __restrict__ W0, const float* __restrict__ W1,
    const float* __restrict__ W2,
    const float* __restrict__ u0, const float* __restrict__ u1,
    const float* __restrict__ u2,
    _Float16* __restrict__ wpack) {
  __shared__ float u[HID], v[HID], red[128];

  float s0 = spectral_sigma(W0, u0, HID, DIN, u, v, red);
  pack_afrags(W0, 1.0f / s0, DIN, HID, wpack + W0_OFF);
  __syncthreads();
  float s1 = spectral_sigma(W1, u1, HID, HID, u, v, red);
  pack_afrags(W1, 1.0f / s1, HID, HID, wpack + W1_OFF);
  __syncthreads();
  float s2 = spectral_sigma(W2, u2, DIN, HID, u, v, red);
  pack_afrags(W2, 1.0f / s2, HID, DIN, wpack + W2_OFF);
}

// ---------------------------------------------------------------------------
// Kernel 2: transposed-flow fused forward + log-det.  A = weights, B = data^T,
// D tiles chain into next-layer B fragments purely in registers.
// ---------------------------------------------------------------------------
__device__ inline v8f wmma16(v16h a, v16h b, v8f c) {
  return __builtin_amdgcn_wmma_f32_16x16x32_f16(
      false, a, false, b, (short)0, c, false, false);
}

__device__ inline float fast_sigmoid(float x) {
  return __builtin_amdgcn_rcpf(1.0f + __expf(-x));   // v_exp + v_rcp
}

// weight A fragment: 16 contiguous halves per lane from a packed blob
__device__ inline v16h load_afrag(const _Float16* blob) {
  const _Float16* p = blob + (threadIdx.x & 31) * 16;
  v8h lo = *(const v8h*)p;
  v8h hi = *(const v8h*)(p + 8);
  return __builtin_shufflevector(lo, hi, 0, 1, 2, 3, 4, 5, 6, 7,
                                 8, 9, 10, 11, 12, 13, 14, 15);
}

// B fragment (K-chunk c) of dataT straight from global f32 rows:
// lane: batch row = lane&15, K = c*32 + (lane>>4)*16 + {0..15} (contiguous)
__device__ inline v16h load_bfrag_global(const float* rowbase, int c) {
  int lane = threadIdx.x & 31;
  const float* p = rowbase + (lane & 15) * DIN + c * 32 + (lane >> 4) * 16;
  v8f f0 = *(const v8f*)p;
  v8f f1 = *(const v8f*)(p + 8);
  v16h r;
  #pragma unroll
  for (int e = 0; e < 8; ++e) {
    r[e] = (_Float16)f0[e];
    r[e + 8] = (_Float16)f1[e];
  }
  return r;
}

// Chain two D-derived f16 octets (feature tiles 2c and 2c+1) into the next
// layer's B fragment via a lane<->lane+16 half-exchange (no LDS arrays).
__device__ inline v16h chain_bfrag(v8h t_even, v8h t_odd) {
  int hi = (threadIdx.x >> 4) & 1;
  v8h send = hi ? t_even : t_odd;
  union { v8h h; int i[4]; } s, r;
  s.h = send;
  #pragma unroll
  for (int q = 0; q < 4; ++q) r.i[q] = __shfl_xor(s.i[q], 16, 32);
  v8h lo = hi ? r.h : t_even;
  v8h hh = hi ? t_odd : r.h;
  return __builtin_shufflevector(lo, hh, 0, 1, 2, 3, 4, 5, 6, 7,
                                 8, 9, 10, 11, 12, 13, 14, 15);
}

#define DS 136   // stride (halves) of d0/d1 rows (128 features + pad)

__global__ __launch_bounds__(256) void iresblock_kernel(
    const float* __restrict__ xg, const float* __restrict__ ldj_in,
    const float* __restrict__ hv,
    const float* __restrict__ b0, const float* __restrict__ b1,
    const float* __restrict__ b2,
    const _Float16* __restrict__ wpack,
    float* __restrict__ zout, float* __restrict__ ldout) {
  __shared__ __align__(16) _Float16 wts[WPACK_HALFS];      // 64 KB, A-frags
  __shared__ float b0s[HID], b1s[HID], b2s[DIN];
  __shared__ __align__(16) _Float16 d0_s[8][16 * DS];      // per wave [row][f]
  __shared__ __align__(16) _Float16 d1_s[8][16 * DS];

  const int tid  = threadIdx.x;
  const int lane = tid & 31;
  const int wid  = tid >> 5;
  const int nrow = lane & 15;        // local batch row
  const int hi16 = lane >> 4;
  const int wrow0 = blockIdx.x * 128 + wid * 16;

  const float* xrow = xg + (size_t)wrow0 * DIN;
  const float* vrow = hv + (size_t)wrow0 * DIN;

  __builtin_prefetch(xrow + (size_t)lane * 16, 0, 1);      // global_prefetch_b8
  __builtin_prefetch(vrow + (size_t)lane * 16, 0, 1);

  // ---- stage packed weights into LDS via async DMA (ASYNCcnt path) --------
  {
    unsigned lds0 = (unsigned)(uintptr_t)&wts[0];
    unsigned long long gbase = (unsigned long long)(uintptr_t)wpack;
    #pragma unroll 4
    for (int i = tid; i < WPACK_HALFS / 8; i += 256) {     // 16B per lane/iter
      unsigned lds_addr = lds0 + (unsigned)i * 16u;
      unsigned voff = (unsigned)i * 16u;
      asm volatile("global_load_async_to_lds_b128 %0, %1, %2"
                   :
                   : "v"(lds_addr), "v"(voff), "s"(gbase)
                   : "memory");
    }
    asm volatile("s_wait_asynccnt 0x0" ::: "memory");
  }
  for (int i = tid; i < HID; i += 256) { b0s[i] = b0[i]; b1s[i] = b1[i]; }
  for (int i = tid; i < DIN; i += 256) { b2s[i] = b2[i]; }

  _Float16* myd0 = &d0_s[wid][0];
  _Float16* myd1 = &d1_s[wid][0];

  __syncthreads();

  const _Float16* w0 = wts + W0_OFF;   // blob (mt,kc): (mt*2+kc)*512
  const _Float16* w1 = wts + W1_OFF;   // blob (mt,kc): (mt*4+kc)*512
  const _Float16* w2 = wts + W2_OFF;   // blob (mt,kc): (mt*4+kc)*512

  // ======================= FORWARD: g(x), z = x + g ========================
  {
    // x as B fragments straight from global (loaded here to keep live ranges
    // short — only the forward pass uses them)
    v16h xf0 = load_bfrag_global(xrow, 0);
    v16h xf1 = load_bfrag_global(xrow, 1);

    v8h hv8[8];
    #pragma unroll
    for (int mt = 0; mt < 8; ++mt) {               // layer 0: D = W0 x^T
      v8f acc = {};
      acc = wmma16(load_afrag(w0 + (mt * 2 + 0) * 512), xf0, acc);
      acc = wmma16(load_afrag(w0 + (mt * 2 + 1) * 512), xf1, acc);
      v8f bb = *(const v8f*)(&b0s[mt * 16 + 8 * hi16]);
      v8h hh, dd;
      #pragma unroll
      for (int j = 0; j < 8; ++j) {
        float pre = acc[j] + bb[j];
        float sg  = fast_sigmoid(pre);
        hh[j] = (_Float16)(pre * sg * (1.0f / 1.1f));
        dd[j] = (_Float16)(sg * (1.0f + pre * (1.0f - sg)) * (1.0f / 1.1f));
      }
      *(v8h*)(myd0 + nrow * DS + mt * 16 + 8 * hi16) = dd;  // one b128 store
      hv8[mt] = hh;
    }
    v16h h1f[4];
    #pragma unroll
    for (int c = 0; c < 4; ++c) h1f[c] = chain_bfrag(hv8[2 * c], hv8[2 * c + 1]);

    #pragma unroll
    for (int mt = 0; mt < 8; ++mt) {               // layer 1
      v8f acc = {};
      #pragma unroll
      for (int kc = 0; kc < 4; ++kc)
        acc = wmma16(load_afrag(w1 + (mt * 4 + kc) * 512), h1f[kc], acc);
      v8f bb = *(const v8f*)(&b1s[mt * 16 + 8 * hi16]);
      v8h hh, dd;
      #pragma unroll
      for (int j = 0; j < 8; ++j) {
        float pre = acc[j] + bb[j];
        float sg  = fast_sigmoid(pre);
        hh[j] = (_Float16)(pre * sg * (1.0f / 1.1f));
        dd[j] = (_Float16)(sg * (1.0f + pre * (1.0f - sg)) * (1.0f / 1.1f));
      }
      *(v8h*)(myd1 + nrow * DS + mt * 16 + 8 * hi16) = dd;
      hv8[mt] = hh;
    }
    v16h h2f[4];
    #pragma unroll
    for (int c = 0; c < 4; ++c) h2f[c] = chain_bfrag(hv8[2 * c], hv8[2 * c + 1]);

    #pragma unroll
    for (int mt = 0; mt < 4; ++mt) {               // layer 2: z = x + g
      v8f acc = {};
      #pragma unroll
      for (int kc = 0; kc < 4; ++kc)
        acc = wmma16(load_afrag(w2 + (mt * 4 + kc) * 512), h2f[kc], acc);
      v8f bb = *(const v8f*)(&b2s[mt * 16 + 8 * hi16]);
      v8f xt = *(const v8f*)(xrow + nrow * DIN + mt * 16 + 8 * hi16);
      v8f zt;
      #pragma unroll
      for (int j = 0; j < 8; ++j) zt[j] = xt[j] + acc[j] + bb[j];
      *(v8f*)(zout + (size_t)(wrow0 + nrow) * DIN + mt * 16 + 8 * hi16) = zt;
    }
  }

  // ============ LOG-DET: w <- J w,  logdet += coef * <w, v> ================
  // v resident in registers for the Hutchinson dot; initial tangent (= v) as
  // B fragments — loaded only now, after the forward live set has died.
  v8f vreg[4];
  #pragma unroll
  for (int mt = 0; mt < 4; ++mt)
    vreg[mt] = *(const v8f*)(vrow + nrow * DIN + mt * 16 + 8 * hi16);
  v16h wfa = load_bfrag_global(vrow, 0);
  v16h wfb = load_bfrag_global(vrow, 1);

  float logacc = 0.0f;
  #pragma unroll 1
  for (int k = 1; k <= NTERMS; ++k) {
    float coef = ((k & 1) ? 1.0f : -1.0f) / (float)k;
    v8h tv[8];
    #pragma unroll
    for (int mt = 0; mt < 8; ++mt) {               // tangent layer 0
      v8f acc = {};
      acc = wmma16(load_afrag(w0 + (mt * 2 + 0) * 512), wfa, acc);
      acc = wmma16(load_afrag(w0 + (mt * 2 + 1) * 512), wfb, acc);
      v8h dd = *(const v8h*)(myd0 + nrow * DS + mt * 16 + 8 * hi16);
      v8h t;
      #pragma unroll
      for (int j = 0; j < 8; ++j) t[j] = (_Float16)(acc[j] * (float)dd[j]);
      tv[mt] = t;
    }
    v16h bf[4];
    #pragma unroll
    for (int c = 0; c < 4; ++c) bf[c] = chain_bfrag(tv[2 * c], tv[2 * c + 1]);

    #pragma unroll
    for (int mt = 0; mt < 8; ++mt) {               // tangent layer 1
      v8f acc = {};
      #pragma unroll
      for (int kc = 0; kc < 4; ++kc)
        acc = wmma16(load_afrag(w1 + (mt * 4 + kc) * 512), bf[kc], acc);
      v8h dd = *(const v8h*)(myd1 + nrow * DS + mt * 16 + 8 * hi16);
      v8h t;
      #pragma unroll
      for (int j = 0; j < 8; ++j) t[j] = (_Float16)(acc[j] * (float)dd[j]);
      tv[mt] = t;
    }
    v16h cf[4];
    #pragma unroll
    for (int c = 0; c < 4; ++c) cf[c] = chain_bfrag(tv[2 * c], tv[2 * c + 1]);

    float part = 0.0f;
    v8h wv8[4];
    #pragma unroll
    for (int mt = 0; mt < 4; ++mt) {               // tangent layer 2 + dot
      v8f acc = {};
      #pragma unroll
      for (int kc = 0; kc < 4; ++kc)
        acc = wmma16(load_afrag(w2 + (mt * 4 + kc) * 512), cf[kc], acc);
      v8f vv = vreg[mt];
      v8h t;
      #pragma unroll
      for (int j = 0; j < 8; ++j) {
        part += acc[j] * vv[j];
        t[j] = (_Float16)acc[j];
      }
      wv8[mt] = t;
    }
    wfa = chain_bfrag(wv8[0], wv8[1]);             // w <- J w (registers only)
    wfb = chain_bfrag(wv8[2], wv8[3]);
    part += __shfl_xor(part, 16, 32);              // combine feature halves
    logacc += coef * part;
  }

  if (lane < 16)
    ldout[wrow0 + nrow] = ldj_in[wrow0 + nrow] + logacc;
}

// ---------------------------------------------------------------------------
extern "C" void kernel_launch(void* const* d_in, const int* in_sizes, int n_in,
                              void* d_out, int out_size, void* d_ws, size_t ws_size,
                              hipStream_t stream) {
  const float* x   = (const float*)d_in[0];
  const float* ldj = (const float*)d_in[1];
  const float* hv  = (const float*)d_in[2];
  const float* W0  = (const float*)d_in[3];
  const float* b0  = (const float*)d_in[4];
  const float* W1  = (const float*)d_in[5];
  const float* b1  = (const float*)d_in[6];
  const float* W2  = (const float*)d_in[7];
  const float* b2  = (const float*)d_in[8];
  const float* u0  = (const float*)d_in[9];
  const float* u1  = (const float*)d_in[10];
  const float* u2  = (const float*)d_in[11];

  float* zout  = (float*)d_out;
  float* ldout = zout + (size_t)BATCH * DIN;
  _Float16* wpack = (_Float16*)d_ws;

  sn_pack_kernel<<<1, 128, 0, stream>>>(W0, W1, W2, u0, u1, u2, wpack);
  iresblock_kernel<<<BATCH / 128, 256, 0, stream>>>(x, ldj, hv, b0, b1, b2,
                                                    wpack, zout, ldout);
}
